// RPN_29437705847001
// MI455X (gfx1250) — matrix operand order repaired and checked by reading, not compiled, not run on hardware
//
#include <hip/hip_runtime.h>
#include <math.h>
#include <stdint.h>

typedef __attribute__((ext_vector_type(16))) _Float16 v16h;
typedef __attribute__((ext_vector_type(8)))  float    v8f;

#define BATCH 8
#define NANCH 39375          // per-batch anchors: 30000 + 7500 + 1875
#define PRE_NMS 1000
#define NMS_TH 0.7f
#define DELTA_CLAMP 4.135166556742356f   // ln(1000/16)

union AFrag { v16h h; uint32_t u[8]; };
union CFrag { v8f  v; float    f[8]; };

typedef __attribute__((address_space(3))) char lds_char;

// ---------------- prep: NCHW f32 -> NHWC f16 ----------------
__global__ __launch_bounds__(256) void prep_x_kernel(const float* __restrict__ src,
                                                     _Float16* __restrict__ dst, int HW) {
  int n = BATCH * HW * 256;
  int i = blockIdx.x * blockDim.x + threadIdx.x;
  if (i >= n) return;
  int c = i & 255;
  int p = (i >> 8) % HW;
  int b = i / (HW * 256);
  dst[i] = (_Float16)src[(b * 256 + c) * HW + p];
}

// ---------------- prep: weights -> per-tap f16 [9][256][256] + head [16][256] ----------------
__global__ __launch_bounds__(256) void prep_w_kernel(const float* __restrict__ cw,
                                                     const float* __restrict__ sw,
                                                     const float* __restrict__ dw,
                                                     _Float16* __restrict__ wh,
                                                     _Float16* __restrict__ hw) {
  int i = blockIdx.x * blockDim.x + threadIdx.x;
  if (i < 9 * 256 * 256) {
    int ci = i & 255; int o = (i >> 8) & 255; int t = i >> 16;
    wh[i] = (_Float16)cw[(o * 256 + ci) * 9 + t];      // OIHW, tap t = ky*3+kx
  } else {
    int j = i - 9 * 256 * 256;
    if (j < 16 * 256) {
      int ci = j & 255; int m = j >> 8;
      float v = 0.f;
      if (m < 3)        v = sw[m * 256 + ci];          // score rows 0..2
      else if (m < 15)  v = dw[(m - 3) * 256 + ci];    // delta rows 3..14
      hw[j] = (_Float16)v;                              // row 15 = pad (0)
    }
  }
}

// ---- fragment loaders (16B-aligned vector accesses) ----
__device__ __forceinline__ void frag_from2(AFrag& f, uint4 lo, uint4 hi) {
  f.u[0] = lo.x; f.u[1] = lo.y; f.u[2] = lo.z; f.u[3] = lo.w;
  f.u[4] = hi.x; f.u[5] = hi.y; f.u[6] = hi.z; f.u[7] = hi.w;
}

// one K-step (t9 = ks>>3 tap, kq = ks&7 cin-block) of the conv GEMM
__device__ __forceinline__ void conv_load_step(int ks, const uint4* __restrict__ whq,
                                               const char* xsb, int aLaneBase,
                                               int nn, int moff, AFrag& A, AFrag& B) {
  int t9 = ks >> 3, kq = ks & 7;
  int dy = (t9 * 11) >> 5;          // t9/3 for t9 in [0,8]
  int dx = t9 - dy * 3;
  int aq = aLaneBase + (t9 << 13) + (kq << 2);     // uint4 units
  frag_from2(A, whq[aq], whq[aq + 2]);
  int boff = ((dy * 18 + nn + dx) << 9) + moff + (kq << 6);   // bytes in xs
  const uint4* lp = (const uint4*)(xsb + boff);
  frag_from2(B, lp[0], lp[2]);
}

// ---------------- fused conv3x3+ReLU (WMMA implicit GEMM) + 1x1 heads ----------------
__global__ __launch_bounds__(512) void conv_head_kernel(
    const _Float16* __restrict__ xh,   // level base, NHWC f16
    const _Float16* __restrict__ wh,   // [9][256][256]
    const _Float16* __restrict__ hw,   // [16][256]
    const float* __restrict__ convb,
    const float* __restrict__ scoreb,
    const float* __restrict__ deltab,
    float* __restrict__ scores,        // [B][NANCH]
    float* __restrict__ deltas,        // [B][NANCH][4]
    int H, int W, int tilesW, int anchorOff)
{
  __shared__ __align__(16) _Float16 xs[3 * 18 * 256];   // 3 rows x 18 cols x 256 cin
  __shared__ __align__(16) _Float16 ts[16 * 256];       // conv out: 16 px x 256 cout

  const int tid  = threadIdx.x;
  const int lane = tid & 31;
  const int wave = tid >> 5;
  const int b    = blockIdx.y;
  const int tile = blockIdx.x;
  const int h    = tile / tilesW;
  const int w0   = (tile % tilesW) << 4;
  const int HW   = H * W;

  const bool interior = (h > 0) && (h + 1 < H) && (w0 > 0) && (w0 + 16 < W);

  if (interior) {
    // CDNA5 async copy: global -> LDS without touching VGPRs (ASYNCcnt-tracked)
    const uint32_t xsLds = (uint32_t)(uintptr_t)(lds_char*)&xs[0];
    for (int t = tid; t < 3 * 18 * 32; t += 512) {       // 1728 chunks of 16B
      int e = t << 3;                                     // halfword index
      int cin = e & 255;
      int rest = e >> 8;
      int ci = rest % 18;
      int dyi = rest / 18;
      const char* gp = (const char*)xh +
          (((size_t)(((b * HW + (h + dyi - 1) * W + (w0 + ci - 1)) << 8) + cin)) << 1);
      uint32_t la = xsLds + ((uint32_t)t << 4);
      asm volatile("global_load_async_to_lds_b128 %0, %1, off"
                   :: "v"(la), "v"(gp) : "memory");
    }
    asm volatile("s_wait_asynccnt 0x0" ::: "memory");
  } else {
    // boundary tiles: zero-padded staging through registers
    uint4* xsq = (uint4*)xs;
    const uint4* xhq = (const uint4*)xh;
    for (int t = tid; t < 3 * 18 * 32; t += 512) {
      int e = t << 3;
      int cin = e & 255;
      int rest = e >> 8;
      int ci = rest % 18;
      int dyi = rest / 18;
      int hin = h + dyi - 1;
      int win = w0 + ci - 1;
      uint4 v; v.x = v.y = v.z = v.w = 0u;
      if (hin >= 0 && hin < H && win >= 0 && win < W)
        v = xhq[(((b * HW + hin * W + win) << 8) + cin) >> 3];
      xsq[t] = v;
    }
  }
  __syncthreads();

  const int coutBase = wave << 4;
  const int mhalf = lane >> 4;            // 0/1
  const int nn = lane & 15;               // A row (=cout) / B col (=pixel)
  const int moff = mhalf << 4;            // byte offset of lane's first 8-k group in LDS rows
  const int aLaneBase = ((coutBase + nn) << 5) + mhalf;   // uint4 index into wh
  const char* xsb = (const char*)xs;
  const uint4* whq = (const uint4*)wh;

  CFrag acc;
#pragma unroll
  for (int r = 0; r < 8; ++r)
    acc.f[r] = convb[coutBase + r + (mhalf << 3)];

  // 72 K-steps (9 taps x 8 cin-blocks), software-pipelined, 2 buffers
  AFrag a0, b0, a1, b1;
  conv_load_step(0, whq, xsb, aLaneBase, nn, moff, a0, b0);
  for (int ks = 0; ks < 72; ks += 2) {
    conv_load_step(ks + 1, whq, xsb, aLaneBase, nn, moff, a1, b1);
    acc.v = __builtin_amdgcn_wmma_f32_16x16x32_f16(false, a0.h, false, b0.h,
                                                   (short)0, acc.v, false, false);
    if (ks + 2 < 72)
      conv_load_step(ks + 2, whq, xsb, aLaneBase, nn, moff, a0, b0);
    acc.v = __builtin_amdgcn_wmma_f32_16x16x32_f16(false, a1.h, false, b1.h,
                                                   (short)0, acc.v, false, false);
  }

  // ReLU, write t tile to LDS as [pixel][cout] (B-matrix layout for head GEMM)
#pragma unroll
  for (int r = 0; r < 8; ++r) {
    int m = r + (mhalf << 3);
    float v = acc.f[r];
    ts[(nn << 8) + coutBase + m] = (_Float16)(v > 0.f ? v : 0.f);
  }
  __syncthreads();

  // head: 16 rows (3 score + 12 delta + pad) x 16 pixels, K=256, B from LDS
  if (wave == 0) {
    const uint4* hwq = (const uint4*)hw;
    const char* tsb = (const char*)ts;
    CFrag hc;
#pragma unroll
    for (int r = 0; r < 8; ++r) hc.f[r] = 0.f;
    for (int kq = 0; kq < 8; ++kq) {
      AFrag a, bm;
      int aq = (nn << 5) + mhalf + (kq << 2);
      frag_from2(a, hwq[aq], hwq[aq + 2]);
      const uint4* lp = (const uint4*)(tsb + ((nn << 9) + moff + (kq << 6)));
      frag_from2(bm, lp[0], lp[2]);
      hc.v = __builtin_amdgcn_wmma_f32_16x16x32_f16(false, a.h, false, bm.h,
                                                    (short)0, hc.v, false, false);
    }
    const int wpix = w0 + nn;
    if (wpix < W) {
      const int pix = h * W + wpix;
      const int abase = b * NANCH + anchorOff + pix * 3;
#pragma unroll
      for (int r = 0; r < 8; ++r) {
        int m = r + (mhalf << 3);
        float v = hc.f[r];
        if (m < 3) {
          scores[abase + m] = v + scoreb[m];
        } else if (m < 15) {
          int j = m - 3;
          deltas[(abase + (j >> 2)) * 4 + (j & 3)] = v + deltab[j];
        }
      }
    }
  }
}

// ---------------- decode + clip + sigmoid (anchors computed analytically) ----------------
__global__ __launch_bounds__(256) void decode_kernel(
    const float* __restrict__ scores, const float* __restrict__ deltas,
    float* __restrict__ boxes, float* __restrict__ probs,
    const int* __restrict__ imh, const int* __restrict__ imw)
{
  int i = blockIdx.x * blockDim.x + threadIdx.x;
  if (i >= BATCH * NANCH) return;
  int b = i / NANCH;
  int t = i - b * NANCH;
  int Wl, stride, local; float size;
  if (t < 30000)      { local = t;         Wl = 100; stride = 8;  size = 64.f;  }
  else if (t < 37500) { local = t - 30000; Wl = 50;  stride = 16; size = 128.f; }
  else                { local = t - 37500; Wl = 25;  stride = 32; size = 256.f; }
  int a   = local % 3;
  int pix = local / 3;
  float x = (float)((pix % Wl) * stride);
  float y = (float)((pix / Wl) * stride);
  float sq = (a == 0) ? 0.70710678118654752f : (a == 1 ? 1.f : 1.41421356237309505f);
  float wa = size / sq;        // anchor width
  float ha = size * sq;        // anchor height
  float dx = deltas[i * 4 + 0], dy = deltas[i * 4 + 1];
  float dw = fminf(deltas[i * 4 + 2], DELTA_CLAMP);
  float dh = fminf(deltas[i * 4 + 3], DELTA_CLAMP);
  float cx = dx * wa + x;
  float cy = dy * ha + y;
  float w  = expf(dw) * wa;
  float hh = expf(dh) * ha;
  float mw = (float)imw[0], mh = (float)imh[0];
  boxes[i * 4 + 0] = fminf(fmaxf(cx - 0.5f * w,  0.f), mw);
  boxes[i * 4 + 1] = fminf(fmaxf(cy - 0.5f * hh, 0.f), mh);
  boxes[i * 4 + 2] = fminf(fmaxf(cx + 0.5f * w,  0.f), mw);
  boxes[i * 4 + 3] = fminf(fmaxf(cy + 0.5f * hh, 0.f), mh);
  probs[i] = 1.f / (1.f + expf(-scores[i]));
}

// ---------------- per-batch stable top-1000 (iterative argmax, LDS bitmask) ----------------
__global__ __launch_bounds__(1024) void topk_kernel(
    const float* __restrict__ probs, const float* __restrict__ boxes,
    float* __restrict__ topb, float* __restrict__ tops)
{
  __shared__ float sv[1024];
  __shared__ int   si[1024];
  __shared__ uint32_t chosen[(NANCH + 31) / 32];
  const int tid = threadIdx.x;
  const int b = blockIdx.x;
  const float* p = probs + b * NANCH;
  for (int i = tid; i < (NANCH + 31) / 32; i += 1024) chosen[i] = 0u;
  __syncthreads();
  for (int k = 0; k < PRE_NMS; ++k) {
    float bv = -1e30f; int bi = NANCH;
    for (int i = tid; i < NANCH; i += 1024) {
      if (chosen[i >> 5] & (1u << (i & 31))) continue;
      float v = p[i];
      if (v > bv || (v == bv && i < bi)) { bv = v; bi = i; }
    }
    sv[tid] = bv; si[tid] = bi;
    __syncthreads();
    for (int s = 512; s > 0; s >>= 1) {
      if (tid < s) {
        float ov = sv[tid + s]; int oi = si[tid + s];
        if (ov > sv[tid] || (ov == sv[tid] && oi < si[tid])) { sv[tid] = ov; si[tid] = oi; }
      }
      __syncthreads();
    }
    if (tid == 0) {
      int idx = si[0];
      chosen[idx >> 5] |= (1u << (idx & 31));
      tops[b * PRE_NMS + k] = sv[0];
      const float* bb = boxes + (size_t)(b * NANCH + idx) * 4;
      float* ob = topb + (size_t)(b * PRE_NMS + k) * 4;
      ob[0] = bb[0]; ob[1] = bb[1]; ob[2] = bb[2]; ob[3] = bb[3];
    }
    __syncthreads();
  }
}

// ---------------- greedy NMS + stable partition to final output ----------------
__global__ __launch_bounds__(256) void nms_kernel(
    const float* __restrict__ topb, const float* __restrict__ tops,
    float* __restrict__ out)
{
  __shared__ float bx[PRE_NMS * 4];
  __shared__ float ar[PRE_NMS];
  __shared__ int   kp[PRE_NMS];
  const int tid = threadIdx.x;
  const int b = blockIdx.x;
  for (int i = tid; i < PRE_NMS; i += 256) {
    float x0 = topb[(b * PRE_NMS + i) * 4 + 0];
    float y0 = topb[(b * PRE_NMS + i) * 4 + 1];
    float x1 = topb[(b * PRE_NMS + i) * 4 + 2];
    float y1 = topb[(b * PRE_NMS + i) * 4 + 3];
    bx[i * 4 + 0] = x0; bx[i * 4 + 1] = y0; bx[i * 4 + 2] = x1; bx[i * 4 + 3] = y1;
    ar[i] = (x1 - x0) * (y1 - y0);
    kp[i] = 1;
  }
  __syncthreads();
  for (int i = 0; i < PRE_NMS - 1; ++i) {
    if (kp[i]) {
      float ax0 = bx[i * 4], ay0 = bx[i * 4 + 1], ax1 = bx[i * 4 + 2], ay1 = bx[i * 4 + 3];
      float aa = ar[i];
      for (int j = i + 1 + tid; j < PRE_NMS; j += 256) {
        if (!kp[j]) continue;
        float iw = fmaxf(fminf(ax1, bx[j * 4 + 2]) - fmaxf(ax0, bx[j * 4 + 0]), 0.f);
        float ih = fmaxf(fminf(ay1, bx[j * 4 + 3]) - fmaxf(ay0, bx[j * 4 + 1]), 0.f);
        float inter = iw * ih;
        float iou = inter / (aa + ar[j] - inter + 1e-9f);
        if (iou > NMS_TH) kp[j] = 0;
      }
    }
    __syncthreads();
  }
  if (tid == 0) {
    float* ob = out + (size_t)b * PRE_NMS * 4;
    float* os = out + (size_t)BATCH * PRE_NMS * 4 + (size_t)b * PRE_NMS;
    int cnt = 0;
    for (int j = 0; j < PRE_NMS; ++j) {
      if (kp[j]) {
        ob[cnt * 4 + 0] = bx[j * 4 + 0]; ob[cnt * 4 + 1] = bx[j * 4 + 1];
        ob[cnt * 4 + 2] = bx[j * 4 + 2]; ob[cnt * 4 + 3] = bx[j * 4 + 3];
        os[cnt] = tops[b * PRE_NMS + j];
        ++cnt;
      }
    }
    for (int j = 0; j < PRE_NMS; ++j) {
      if (!kp[j]) {
        ob[cnt * 4 + 0] = bx[j * 4 + 0]; ob[cnt * 4 + 1] = bx[j * 4 + 1];
        ob[cnt * 4 + 2] = bx[j * 4 + 2]; ob[cnt * 4 + 3] = bx[j * 4 + 3];
        os[cnt] = -1.f;   // suppressed scores become -1 after final top_k
        ++cnt;
      }
    }
  }
}

extern "C" void kernel_launch(void* const* d_in, const int* in_sizes, int n_in,
                              void* d_out, int out_size, void* d_ws, size_t ws_size,
                              hipStream_t stream) {
  (void)in_sizes; (void)n_in; (void)out_size;
  const float* feat0   = (const float*)d_in[0];
  const float* feat1   = (const float*)d_in[1];
  const float* feat2   = (const float*)d_in[2];
  const float* conv_w  = (const float*)d_in[3];
  const float* conv_b  = (const float*)d_in[4];
  const float* score_w = (const float*)d_in[5];
  const float* score_b = (const float*)d_in[6];
  const float* delta_w = (const float*)d_in[7];
  const float* delta_b = (const float*)d_in[8];
  const int*   image_h = (const int*)d_in[9];
  const int*   image_w = (const int*)d_in[10];
  float* out = (float*)d_out;

  // workspace layout
  char* w = (char*)d_ws;
  size_t off = 0;
  auto alloc = [&](size_t bytes) { void* p = w + off; off = (off + bytes + 255) & ~(size_t)255; return p; };
  _Float16* XH  = (_Float16*)alloc(26880000ull * 2);  // NHWC f16 features (all levels)
  _Float16* WH  = (_Float16*)alloc(589824ull * 2);    // [9][256][256]
  _Float16* HWm = (_Float16*)alloc(4096ull * 2);      // [16][256]
  float* SC   = (float*)alloc(315000ull * 4);         // scores  [B][NANCH]
  float* DL   = (float*)alloc(1260000ull * 4);        // deltas  [B][NANCH][4]
  float* BX   = (float*)alloc(1260000ull * 4);        // boxes   [B][NANCH][4]
  float* PR   = (float*)alloc(315000ull * 4);         // probs   [B][NANCH]
  float* TOPB = (float*)alloc(32000ull * 4);          // [B][1000][4]
  float* TOPS = (float*)alloc(8000ull * 4);           // [B][1000]
  if (off > ws_size) return;

  // prep
  { int n = BATCH * 10000 * 256; prep_x_kernel<<<(n + 255) / 256, 256, 0, stream>>>(feat0, XH, 10000); }
  { int n = BATCH * 2500  * 256; prep_x_kernel<<<(n + 255) / 256, 256, 0, stream>>>(feat1, XH + 20480000, 2500); }
  { int n = BATCH * 625   * 256; prep_x_kernel<<<(n + 255) / 256, 256, 0, stream>>>(feat2, XH + 25600000, 625); }
  { int n = 9 * 256 * 256 + 16 * 256;
    prep_w_kernel<<<(n + 255) / 256, 256, 0, stream>>>(conv_w, score_w, delta_w, WH, HWm); }

  // fused conv + heads, one launch per level
  conv_head_kernel<<<dim3(7 * 100, BATCH), 512, 0, stream>>>(
      XH,            WH, HWm, conv_b, score_b, delta_b, SC, DL, 100, 100, 7, 0);
  conv_head_kernel<<<dim3(4 * 50,  BATCH), 512, 0, stream>>>(
      XH + 20480000, WH, HWm, conv_b, score_b, delta_b, SC, DL, 50,  50,  4, 30000);
  conv_head_kernel<<<dim3(2 * 25,  BATCH), 512, 0, stream>>>(
      XH + 25600000, WH, HWm, conv_b, score_b, delta_b, SC, DL, 25,  25,  2, 37500);

  // decode + sigmoid
  { int n = BATCH * NANCH;
    decode_kernel<<<(n + 255) / 256, 256, 0, stream>>>(SC, DL, BX, PR, image_h, image_w); }

  // per-batch top-1000, NMS, finalize
  topk_kernel<<<BATCH, 1024, 0, stream>>>(PR, BX, TOPB, TOPS);
  nms_kernel<<<BATCH, 256, 0, stream>>>(TOPB, TOPS, out);
}
